// NAOLayer_14637248545384
// MI455X (gfx1250) — matrix operand order repaired and checked by reading, not compiled, not run on hardware
//
#include <hip/hip_runtime.h>

#define B_   4
#define N_   8192
#define D_   1024
#define DK_  128
#define TN1  32          // rows of J per block in pass 1
#define TN3  32          // rows per block in output pass
#define LDJ  (D_ + 8)    // padded LDS strides (avoid bank conflicts)
#define LDW  (128 + 8)
#define LDKT (TN1 + 8)
#define LDM  (128 + 8)   // matches TDM pad: 64-DWORD interval + 4-DWORD pad
#define LDO  (D_ + 8)
#define LDQ  (DK_ + 8)

typedef __bf16 bf16_t;
typedef __attribute__((ext_vector_type(16))) __bf16 bf16x16;
typedef __attribute__((ext_vector_type(8)))  float  floatx8;
typedef __attribute__((ext_vector_type(4)))  unsigned int uint32x4;
typedef __attribute__((ext_vector_type(8)))  int  int32x8;
typedef __attribute__((ext_vector_type(4)))  int  int32x4;

static __device__ inline bf16_t f2bf(float f) {
    unsigned u = __builtin_bit_cast(unsigned, f);
    unsigned r = (u + 0x7FFFu + ((u >> 16) & 1u)) >> 16;   // round-to-nearest-even
    unsigned short s = (unsigned short)r;
    return __builtin_bit_cast(bf16_t, s);
}
static __device__ inline float bf2f(bf16_t b) {
    unsigned short s = __builtin_bit_cast(unsigned short, b);
    unsigned u = ((unsigned)s) << 16;
    return __builtin_bit_cast(float, u);
}

static __device__ inline floatx8 wmma_bf16(bf16x16 a, bf16x16 b, floatx8 c) {
    return __builtin_amdgcn_wmma_f32_16x16x32_bf16(
        /*neg_a=*/false, a, /*neg_b=*/false, b,
        /*c_mod=*/(short)0, c, /*reuse_a=*/false, /*reuse_b=*/false);
}

// ---- Tensor Data Mover: 2D bf16 tile load with LDS row padding -------------
// D# packing per CDNA5 ISA ch.8 (groups 0/1; groups 2/3 zero => 2D tensor).
// pad: interval = 64 DWORDs (code 5) / amount = 4 DWORDs (code 3) => LDS row
// stride of 128+8 bf16 elements, matching LDM/LDQ.
// 6-arg builtin variant (clang-23 / therock-10.0 headers).
static __device__ inline void tdm_load_2d_bf16(unsigned lds_byte_off,
                                               const void* gaddr,
                                               unsigned tensor_d0, unsigned tensor_d1,
                                               unsigned tile_d0, unsigned tile_d1,
                                               unsigned long long stride_d0) {
    unsigned long long ga = (unsigned long long)(size_t)gaddr;
    uint32x4 g0 = {};
    g0[0] = 1u;                                            // count=1 (user D#)
    g0[1] = lds_byte_off;                                  // LDS byte address
    g0[2] = (unsigned)(ga & 0xFFFFFFFFu);                  // global_addr[31:0]
    g0[3] = (unsigned)((ga >> 32) & 0x01FFFFFFu) | (2u << 30);  // addr[56:32] | type=2
    int32x8 g1 = {};
    g1[0] = (int)((1u << 16)        // data_size = 1 (2 bytes)
                | (1u << 20)        // pad_enable
                | (5u << 22)        // pad_interval: 64 DWORDs
                | (3u << 25));      // pad_amount: 4 DWORDs
    g1[1] = (int)((tensor_d0 & 0xFFFFu) << 16);                        // dim0 lo16
    g1[2] = (int)(((tensor_d0 >> 16) & 0xFFFFu) | ((tensor_d1 & 0xFFFFu) << 16));
    g1[3] = (int)(((tensor_d1 >> 16) & 0xFFFFu) | ((tile_d0 & 0xFFFFu) << 16));
    g1[4] = (int)(tile_d1 & 0xFFFFu);                                  // tile_dim1, tile_dim2=0
    g1[5] = (int)(unsigned)(stride_d0 & 0xFFFFFFFFull);                // dim0_stride lo32
    g1[6] = (int)(unsigned)((stride_d0 >> 32) & 0xFFFFull);            // dim0_stride hi16
    g1[7] = 0;
    int32x4 z4 = {};
    int32x8 z8 = {};
    __builtin_amdgcn_tensor_load_to_lds(g0, g1, z4, z4, z8, 0);
}

// A fragment 16(M)x32(K), source row-major M x K with stride ld (elements).
static __device__ inline bf16x16 load_a_frag(const bf16_t* s, int ld) {
    int lane = threadIdx.x & 31;
    int row  = lane & 15;
    int half = lane >> 4;
    const bf16_t* p = s + row * ld;
    bf16x16 a;
#pragma unroll
    for (int j = 0; j < 8; ++j) a[j] = p[half * 16 + j];
#pragma unroll
    for (int j = 0; j < 8; ++j) a[8 + j] = p[half * 16 + 8 + j];
    return a;
}

// B fragment 32(K)x16(N), source stored N-major: s[n*ld + k] (K contiguous).
static __device__ inline bf16x16 load_b_frag_kcontig(const bf16_t* s, int ld) {
    int lane  = threadIdx.x & 31;
    int n     = lane & 15;
    int kbase = (lane >> 4) * 16;
    const bf16_t* p = s + n * ld + kbase;
    bf16x16 b;
#pragma unroll
    for (int j = 0; j < 16; ++j) b[j] = p[j];
    return b;
}

// B fragment 32(K)x16(N), source stored K-major: s[k*ld + n].
static __device__ inline bf16x16 load_b_frag_ncontig(const bf16_t* s, int ld) {
    int lane  = threadIdx.x & 31;
    int n     = lane & 15;
    int kbase = (lane >> 4) * 16;
    bf16x16 b;
#pragma unroll
    for (int j = 0; j < 16; ++j) b[j] = s[(kbase + j) * ld + n];
    return b;
}

__global__ void k_zero(float* p, int n) {
    int i = blockIdx.x * 256 + threadIdx.x;
    if (i < n) p[i] = 0.0f;
}

// Pass 1: Q = J W_Q^T (to workspace, bf16), M += K^T J (fp32 atomics).
__global__ __launch_bounds__(256) void k_qkm(const float* __restrict__ J,
                                             const float* __restrict__ WQ,
                                             const float* __restrict__ WK,
                                             bf16_t* __restrict__ Qbf,
                                             float* __restrict__ Mf) {
    __shared__ bf16_t sJ[TN1 * LDJ];
    __shared__ bf16_t sWq[128 * LDW];
    __shared__ bf16_t sWk[128 * LDW];
    __shared__ bf16_t sKt[DK_ * LDKT];

    int b   = blockIdx.x / (N_ / TN1);
    int nt  = blockIdx.x % (N_ / TN1);
    int n0  = nt * TN1;
    int tid = threadIdx.x;
    int wave = tid >> 5;
    int lane = tid & 31;
    int cn   = lane & 15;
    int rbase = (lane < 16) ? 0 : 8;

    const float* Jb = J + ((size_t)b * N_ + n0) * D_;

    // stage 32x1024 J chunk as bf16 (float4 global reads)
    for (int e = tid; e < TN1 * D_ / 4; e += 256) {
        int r = e / (D_ / 4), c4 = e % (D_ / 4);
        float4 v = ((const float4*)(Jb + (size_t)r * D_))[c4];
        bf16_t* d = &sJ[r * LDJ + c4 * 4];
        d[0] = f2bf(v.x); d[1] = f2bf(v.y); d[2] = f2bf(v.z); d[3] = f2bf(v.w);
    }
    __syncthreads();

    // GEMM1: wave owns output column-tile `wave` for Q and K, row-tiles 0/1.
    floatx8 accQ0 = {}, accQ1 = {}, accK0 = {}, accK1 = {};
    for (int dt = 0; dt < 8; ++dt) {
        for (int e = tid; e < 128 * 32; e += 256) {   // 128x128 fp32 per matrix
            int r = e / 32, c4 = e % 32;
            float4 q = ((const float4*)(WQ + (size_t)r * D_ + dt * 128))[c4];
            float4 k = ((const float4*)(WK + (size_t)r * D_ + dt * 128))[c4];
            bf16_t* dq = &sWq[r * LDW + c4 * 4];
            dq[0] = f2bf(q.x); dq[1] = f2bf(q.y); dq[2] = f2bf(q.z); dq[3] = f2bf(q.w);
            bf16_t* dk = &sWk[r * LDW + c4 * 4];
            dk[0] = f2bf(k.x); dk[1] = f2bf(k.y); dk[2] = f2bf(k.z); dk[3] = f2bf(k.w);
        }
        __syncthreads();
#pragma unroll
        for (int ks = 0; ks < 4; ++ks) {
            int dof = dt * 128 + ks * 32;
            bf16x16 a0 = load_a_frag(&sJ[0 * LDJ + dof], LDJ);
            bf16x16 a1 = load_a_frag(&sJ[16 * LDJ + dof], LDJ);
            bf16x16 bq = load_b_frag_kcontig(&sWq[(wave * 16) * LDW + ks * 32], LDW);
            bf16x16 bk = load_b_frag_kcontig(&sWk[(wave * 16) * LDW + ks * 32], LDW);
            accQ0 = wmma_bf16(a0, bq, accQ0);
            accQ1 = wmma_bf16(a1, bq, accQ1);
            accK0 = wmma_bf16(a0, bk, accK0);
            accK1 = wmma_bf16(a1, bk, accK1);
        }
        __syncthreads();
    }

    // write Q chunk (bf16) to workspace; transpose K chunk into LDS
    bf16_t* Qg = Qbf + ((size_t)b * N_ + n0) * DK_ + wave * 16 + cn;
    bf16_t* kt = &sKt[(wave * 16 + cn) * LDKT];
#pragma unroll
    for (int r = 0; r < 8; ++r) {
        int m = r + rbase;
        Qg[(size_t)m * DK_]        = f2bf(accQ0[r]);
        Qg[(size_t)(16 + m) * DK_] = f2bf(accQ1[r]);
        kt[m]      = f2bf(accK0[r]);
        kt[16 + m] = f2bf(accK1[r]);
    }
    __syncthreads();

    // GEMM2: M_part(128x1024) = Kt(128x32) @ J(32x1024); wave owns row-tile `wave`.
    bf16x16 aK = load_a_frag(&sKt[(wave * 16) * LDKT], LDKT);
    float* Mb = Mf + ((size_t)b * DK_ + wave * 16) * D_;
    for (int ct = 0; ct < 64; ++ct) {
        bf16x16 bj = load_b_frag_ncontig(&sJ[ct * 16], LDJ);
        floatx8 acc = {};
        acc = wmma_bf16(aK, bj, acc);
        float* Mp = Mb + ct * 16 + cn;
#pragma unroll
        for (int r = 0; r < 8; ++r)
            unsafeAtomicAdd(&Mp[(size_t)(r + rbase) * D_], acc[r]);
    }
}

// fold 1/sqrt(d_k) while converting M to bf16
__global__ void k_mconv(const float* __restrict__ Mf, bf16_t* __restrict__ Mbf) {
    int i = blockIdx.x * 256 + threadIdx.x;
    if (i < B_ * DK_ * D_) Mbf[i] = f2bf(Mf[i] * 0.08838834764831845f);
}

// Pass 2: J_new = Q*M + J, fused LayerNorm, fp32 out.
// Q tile and M tiles are bf16 in workspace -> staged via TDM (tensor_load_to_lds)
// with double-buffered M tiles (TENSORcnt pipeline).
__global__ __launch_bounds__(256) void k_out(const float* __restrict__ J,
                                             const bf16_t* __restrict__ Qbf,
                                             const bf16_t* __restrict__ Mbf,
                                             const float* __restrict__ gamma,
                                             const float* __restrict__ beta,
                                             float* __restrict__ out) {
    __shared__ bf16_t sQ[TN3 * LDQ];
    __shared__ bf16_t sM[2][DK_ * LDM];
    __shared__ bf16_t sO[TN3 * LDO];
    __shared__ float  rowsum[TN3], rowsq[TN3], rmu[TN3], rsg[TN3];

    int b   = blockIdx.x / (N_ / TN3);
    int nt  = blockIdx.x % (N_ / TN3);
    int n0  = nt * TN3;
    int tid = threadIdx.x;
    int wave = tid >> 5;
    int lane = tid & 31;
    int cn   = lane & 15;
    int rbase = (lane < 16) ? 0 : 8;

    const bf16_t* Qg = Qbf + ((size_t)b * N_ + n0) * DK_;
    const bf16_t* Mg = Mbf + (size_t)b * DK_ * D_;

    if (tid < TN3) { rowsum[tid] = 0.0f; rowsq[tid] = 0.0f; }

    // wave 0 drives the TDM: Q tile (TN3 x 128) + first M tile (128 x 128)
    if (wave == 0) {
        tdm_load_2d_bf16((unsigned)(size_t)(void*)&sQ[0], Qg,
                         /*tensor_d0=*/DK_, /*tensor_d1=*/N_ - n0,
                         /*tile_d0=*/DK_, /*tile_d1=*/TN3, /*stride_d0=*/DK_);
        tdm_load_2d_bf16((unsigned)(size_t)(void*)&sM[0][0], Mg,
                         /*tensor_d0=*/D_, /*tensor_d1=*/DK_,
                         /*tile_d0=*/128, /*tile_d1=*/DK_, /*stride_d0=*/D_);
    }

    for (int dt = 0; dt < 8; ++dt) {
        int cur = dt & 1;
        if (wave == 0) __builtin_amdgcn_s_wait_tensorcnt(0);
        __syncthreads();                       // sQ + sM[cur] visible to all waves
        // prefetch next M tile into the buffer whose readers retired at the
        // barrier above (read during iteration dt-1)
        if (wave == 0 && dt < 7) {
            tdm_load_2d_bf16((unsigned)(size_t)(void*)&sM[1 - cur][0],
                             Mg + (dt + 1) * 128,
                             /*tensor_d0=*/D_, /*tensor_d1=*/DK_,
                             /*tile_d0=*/128, /*tile_d1=*/DK_, /*stride_d0=*/D_);
        }

        floatx8 acc0 = {}, acc1 = {};
#pragma unroll
        for (int ks = 0; ks < 4; ++ks) {
            bf16x16 a0 = load_a_frag(&sQ[0 * LDQ + ks * 32], LDQ);
            bf16x16 a1 = load_a_frag(&sQ[16 * LDQ + ks * 32], LDQ);
            bf16x16 bm = load_b_frag_ncontig(&sM[cur][(ks * 32) * LDM + wave * 16], LDM);
            acc0 = wmma_bf16(a0, bm, acc0);
            acc1 = wmma_bf16(a1, bm, acc1);
        }
        // residual add, stash J_new tile (bf16) in LDS
        const float* Jg = J + ((size_t)b * N_ + n0) * D_ + dt * 128 + wave * 16 + cn;
        int co = dt * 128 + wave * 16 + cn;
#pragma unroll
        for (int r = 0; r < 8; ++r) {
            int m0 = r + rbase, m1 = 16 + r + rbase;
            sO[m0 * LDO + co] = f2bf(acc0[r] + Jg[(size_t)m0 * D_]);
            sO[m1 * LDO + co] = f2bf(acc1[r] + Jg[(size_t)m1 * D_]);
        }
    }
    __syncthreads();                           // all sO tiles written

    // LayerNorm: partial sums (8 segments of 128 per row), LDS float atomics
    {
        int r = tid & 31, seg = tid >> 5;
        float s = 0.0f, sq = 0.0f;
        const bf16_t* p = &sO[r * LDO + seg * 128];
#pragma unroll 4
        for (int i = 0; i < 128; ++i) { float v = bf2f(p[i]); s += v; sq += v * v; }
        atomicAdd(&rowsum[r], s);
        atomicAdd(&rowsq[r], sq);
    }
    __syncthreads();
    if (tid < TN3) {
        float mu  = rowsum[tid] * (1.0f / 1024.0f);
        float var = rowsq[tid] * (1.0f / 1024.0f) - mu * mu;
        rmu[tid] = mu;
        rsg[tid] = rsqrtf(var + 1e-5f);
    }
    __syncthreads();

    float* Og = out + ((size_t)b * N_ + n0) * D_;
    for (int e = tid; e < TN3 * D_; e += 256) {
        int r = e >> 10, c = e & 1023;
        float v = bf2f(sO[r * LDO + c]);
        Og[e] = (v - rmu[r]) * rsg[r] * gamma[c] + beta[c];
    }
}

extern "C" void kernel_launch(void* const* d_in, const int* in_sizes, int n_in,
                              void* d_out, int out_size, void* d_ws, size_t ws_size,
                              hipStream_t stream) {
    const float* J     = (const float*)d_in[0];
    const float* WQ    = (const float*)d_in[1];
    const float* WK    = (const float*)d_in[2];
    const float* gamma = (const float*)d_in[3];
    const float* beta  = (const float*)d_in[4];
    float* out = (float*)d_out;

    char* ws = (char*)d_ws;
    bf16_t* Qbf = (bf16_t*)ws;                                   // B*N*DK bf16 = 8 MB
    float*  Mf  = (float*)(ws + (size_t)8 * 1024 * 1024);        // B*DK*D fp32 = 2 MB
    bf16_t* Mbf = (bf16_t*)(ws + (size_t)10 * 1024 * 1024);      // B*DK*D bf16 = 1 MB

    const int mElems = B_ * DK_ * D_;
    k_zero<<<(mElems + 255) / 256, 256, 0, stream>>>(Mf, mElems);
    k_qkm<<<B_ * (N_ / TN1), 256, 0, stream>>>(J, WQ, WK, Qbf, Mf);
    k_mconv<<<(mElems + 255) / 256, 256, 0, stream>>>(Mf, Mbf);
    k_out<<<B_ * (N_ / TN3), 256, 0, stream>>>(J, Qbf, Mbf, gamma, beta, out);
}
